// MultiScaleSparseSelfAttention_85220741087703
// MI455X (gfx1250) — matrix operand order repaired and checked by reading, not compiled, not run on hardware
//
#include <hip/hip_runtime.h>
#include <hip/hip_bf16.h>

typedef __attribute__((ext_vector_type(16))) __bf16 v16bf;
typedef __attribute__((ext_vector_type(8)))  __bf16 v8bf;
typedef __attribute__((ext_vector_type(8)))  float  v8f;

// explicit address spaces so global traffic lowers to global_load/store_* (separate
// LOADcnt/STOREcnt) instead of flat_* (which also ties up DScnt), and LDS to ds_*.
typedef __attribute__((address_space(1))) __bf16  g_bf;
typedef __attribute__((address_space(1))) float   g_f32;
typedef __attribute__((address_space(1))) v8bf    g_v8bf;
typedef __attribute__((address_space(1))) v16bf   g_v16bf;
typedef __attribute__((address_space(1))) v8f     g_v8f;
typedef __attribute__((address_space(3))) __bf16  s_bf;
typedef __attribute__((address_space(3))) v8bf    s_v8bf;
typedef __attribute__((address_space(3))) v16bf   s_v16bf;

#define WMMA_BF16(a, b, c) \
  __builtin_amdgcn_wmma_f32_16x16x32_bf16(false, (a), false, (b), (short)0, (c), false, false)

// ---- fragment loaders -------------------------------------------------------
// A (16x32, 16-bit): lane L holds row L%16; vector elems 0-15 = K[c0..c0+16),
// c0 = (L<16 ? 0 : 16).  Caller passes row ptr + c0.
__device__ __forceinline__ v16bf load_afrag_g(const g_bf* p) {
  v8bf lo = *(const g_v8bf*)(p);
  v8bf hh = *(const g_v8bf*)(p + 8);
  v16bf r;
#pragma unroll
  for (int i = 0; i < 8; ++i) { r[i] = lo[i]; r[i + 8] = hh[i]; }
  return r;
}
__device__ __forceinline__ v16bf load_afrag_s(const s_bf* p) {
  v8bf lo = *(const s_v8bf*)(p);
  v8bf hh = *(const s_v8bf*)(p + 8);
  v16bf r;
#pragma unroll
  for (int i = 0; i < 8; ++i) { r[i] = lo[i]; r[i + 8] = hh[i]; }
  return r;
}
__device__ __forceinline__ v16bf load_afrag_gf32(const g_f32* p) {
  v8f lo = *(const g_v8f*)(p);
  v8f hh = *(const g_v8f*)(p + 8);
  v16bf r;
#pragma unroll
  for (int i = 0; i < 8; ++i) { r[i] = (__bf16)lo[i]; r[i + 8] = (__bf16)hh[i]; }
  return r;
}
// B (32x16, 16-bit): lane L holds col L%16; vector elems = contiguous K run of 16
// starting at K = (L<16 ? 0 : 16).  Caller passes ptr to the start of that run.
__device__ __forceinline__ v16bf load_bfrag_g(const g_bf* p) {
  return *(const g_v16bf*)p;   // 32B, aligned by construction
}
__device__ __forceinline__ v16bf load_bfrag_gf32(const g_f32* p) {
  v8f lo = *(const g_v8f*)(p);
  v8f hh = *(const g_v8f*)(p + 8);
  v16bf r;
#pragma unroll
  for (int i = 0; i < 8; ++i) { r[i] = (__bf16)lo[i]; r[i + 8] = (__bf16)hh[i]; }
  return r;
}

// ---- Stage 1: Q/K/V projections  (xf[b,n,c] @ W^T + bias) -------------------
// One wave computes a 16(n) x 64(o) tile; o-tile == head (d = o%64).
// Q,K -> bf16 [sb, h, n, d]; V -> f32 same layout.
__global__ void proj_kernel(
    const float* __restrict__ x0, const float* __restrict__ x1,
    const float* __restrict__ Wq0, const float* __restrict__ bq0,
    const float* __restrict__ Wk0, const float* __restrict__ bk0,
    const float* __restrict__ Wv0, const float* __restrict__ bv0,
    const float* __restrict__ Wq1, const float* __restrict__ bq1,
    const float* __restrict__ Wk1, const float* __restrict__ bk1,
    const float* __restrict__ Wv1, const float* __restrict__ bv1,
    __bf16* __restrict__ Qws, __bf16* __restrict__ Kws, float* __restrict__ Vws)
{
  const int lane = threadIdx.x & 31;
  const int wave = threadIdx.x >> 5;
  const int gw   = blockIdx.x * 2 + wave;          // 3072 waves total
  const int mat  = gw >> 8;                        // 0..11  = (scale,b,proj)
  const int tile = gw & 255;
  const int nt = tile & 63, ot = tile >> 6;
  const int sc = mat / 6, rem = mat % 6, bb = rem / 3, pj = rem % 3;

  const float* Wtab[6] = { Wq0, Wk0, Wv0, Wq1, Wk1, Wv1 };
  const float* Btab[6] = { bq0, bk0, bv0, bq1, bk1, bv1 };
  const g_f32* W    = (const g_f32*)Wtab[sc * 3 + pj];
  const g_f32* bias = (const g_f32*)Btab[sc * 3 + pj];
  const g_f32* xf   = (const g_f32*)(sc ? x1 : x0) + (size_t)bb * 1024 * 256;

  const int lr = lane & 15, hi = lane >> 4;
  const int row = nt * 16 + lr;

  v8f acc[4] = { v8f{}, v8f{}, v8f{}, v8f{} };
#pragma unroll
  for (int cc = 0; cc < 8; ++cc) {                 // K-dim (C=256) in chunks of 32
    v16bf af = load_afrag_gf32(xf + row * 256 + cc * 32 + hi * 16);
#pragma unroll
    for (int t = 0; t < 4; ++t) {
      v16bf bf = load_bfrag_gf32(W + (ot * 64 + t * 16 + lr) * 256 + cc * 32 + hi * 16);
      acc[t] = WMMA_BF16(af, bf, acc[t]);
    }
  }

  const size_t base = (size_t)((sc * 2 + bb) * 4 + ot) * 65536;  // [sb,h] * N*d
  g_bf*  Qg = (g_bf*)Qws;
  g_bf*  Kg = (g_bf*)Kws;
  g_f32* Vg = (g_f32*)Vws;
#pragma unroll
  for (int t = 0; t < 4; ++t) {
    const int d = t * 16 + lr;
    const float bv = bias[ot * 64 + d];
#pragma unroll
    for (int r = 0; r < 8; ++r) {
      const int n = nt * 16 + r + 8 * hi;
      const float val = acc[t][r] + bv;
      if (pj == 0)      Qg[base + n * 64 + d] = (__bf16)val;
      else if (pj == 1) Kg[base + n * 64 + d] = (__bf16)val;
      else              Vg[base + n * 64 + d] = val;
    }
  }
}

// ---- Stage 2: VsumT[sb,h,d,m] = bf16( sum_{s in {-4,-2,0,2,4}} V[sb,h,(m+s)%N,d] )
__global__ void vsum_kernel(const float* __restrict__ Vws, __bf16* __restrict__ VT)
{
  const int idx = blockIdx.x * 256 + threadIdx.x;  // 1,048,576 elements
  const int m = idx & 1023;
  const int d = (idx >> 10) & 63;
  const int sh = idx >> 16;                        // sb*4 + h
  const g_f32* vb = (const g_f32*)Vws + (size_t)sh * 65536;
  float s = 0.f;
#pragma unroll
  for (int k = -4; k <= 4; k += 2) s += vb[((m + k) & 1023) * 64 + d];
  ((g_bf*)VT)[(size_t)sh * 65536 + d * 1024 + m] = (__bf16)s;
}

// ---- Stage 3: attention  out[h] = sum_g w(h,g) softmax(Qh Kg^T /16) @ Vsum_{(2h-g)%4}
// grid 256: b(2) x h(4) x rowtile32(32).  block 64 = 2 waves, each wave: 16 q rows.
// LDS: per-wave P tile 16x1024 bf16 (32KB) -> 64KB dynamic.
__global__ void attn_kernel(const __bf16* __restrict__ Qws,
                            const __bf16* __restrict__ Kws,
                            const __bf16* __restrict__ VT,
                            float* __restrict__ out)
{
  extern __shared__ __bf16 smem[];
  const int lane = threadIdx.x & 31, wave = threadIdx.x >> 5;
  s_bf* P = (s_bf*)smem + (size_t)wave * 16 * 1024;
  const int lr = lane & 15, hi = lane >> 4;

  const int bi = blockIdx.x >> 7;
  const int h  = (blockIdx.x >> 5) & 3;
  const int rt = blockIdx.x & 31;
  const int row0 = rt * 32 + wave * 16;
  const int qrow = row0 + lr;

  v8f acc[4] = { v8f{}, v8f{}, v8f{}, v8f{} };

  for (int sc = 0; sc < 2; ++sc) {
    const int sb = sc * 2 + bi;
    const g_bf* Qb = (const g_bf*)Qws + (size_t)(sb * 4 + h) * 65536;
    const v16bf qf0 = load_afrag_g(Qb + qrow * 64 + hi * 16);
    const v16bf qf1 = load_afrag_g(Qb + qrow * 64 + 32 + hi * 16);

    for (int g = 0; g < 4; ++g) {
      const g_bf* Kb = (const g_bf*)Kws + (size_t)(sb * 4 + g) * 65536;

      // --- S = exp(Q K^T / 16): write bf16 probs to LDS, keep per-lane row sums
      float rs[8];
#pragma unroll
      for (int r = 0; r < 8; ++r) rs[r] = 0.f;

      for (int mt = 0; mt < 64; ++mt) {
        const int krow = mt * 16 + lr;
        v16bf kf0 = load_bfrag_g(Kb + krow * 64 + hi * 16);
        v16bf kf1 = load_bfrag_g(Kb + krow * 64 + 32 + hi * 16);
        v8f c = v8f{};
        c = WMMA_BF16(qf0, kf0, c);
        c = WMMA_BF16(qf1, kf1, c);
#pragma unroll
        for (int r = 0; r < 8; ++r) {
          const float e = __expf(c[r] * 0.0625f);   // 1/sqrt(OUT) = 1/16
          rs[r] += e;
          P[(8 * hi + r) * 1024 + mt * 16 + lr] = (__bf16)e;
        }
      }
      // butterfly-sum row sums across the 16-lane half (wave32 shuffles)
#pragma unroll
      for (int r = 0; r < 8; ++r) {
        float v = rs[r];
        v += __shfl_xor(v, 1, 32);
        v += __shfl_xor(v, 2, 32);
        v += __shfl_xor(v, 4, 32);
        v += __shfl_xor(v, 8, 32);
        rs[r] = v;
      }
      __syncthreads();

      // --- cg = P @ Vsum_{(2h-g)%4}   (VT stored [d, m] -> contiguous B-frags)
      const int hv = (2 * h - g + 4) & 3;
      const g_bf* Vb = (const g_bf*)VT + (size_t)(sb * 4 + hv) * 65536;
      v8f cg[4] = { v8f{}, v8f{}, v8f{}, v8f{} };
      for (int mc = 0; mc < 32; ++mc) {
        v16bf pf = load_afrag_s(P + lr * 1024 + mc * 32 + hi * 16);
#pragma unroll
        for (int dt = 0; dt < 4; ++dt) {
          v16bf vf = load_bfrag_g(Vb + (dt * 16 + lr) * 1024 + mc * 32 + hi * 16);
          cg[dt] = WMMA_BF16(pf, vf, cg[dt]);
        }
      }

      // --- fold with softmax normalization and head-pair weight
      const float wgt = (g == ((h + 2) & 3)) ? 2.0f : 1.0f;
      float f[8];
#pragma unroll
      for (int r = 0; r < 8; ++r) f[r] = wgt / rs[r];
#pragma unroll
      for (int dt = 0; dt < 4; ++dt)
#pragma unroll
        for (int r = 0; r < 8; ++r) acc[dt][r] += cg[dt][r] * f[r];

      __syncthreads();
    }
  }

  // store: out[b, n, h*64 + d], summed over both scales
  g_f32* og = (g_f32*)out;
#pragma unroll
  for (int dt = 0; dt < 4; ++dt)
#pragma unroll
    for (int r = 0; r < 8; ++r) {
      const int n = row0 + r + 8 * hi;
      const int o = h * 64 + dt * 16 + lr;
      og[((size_t)bi * 1024 + n) * 256 + o] = acc[dt][r];
    }
}

// ---- host ------------------------------------------------------------------
extern "C" void kernel_launch(void* const* d_in, const int* in_sizes, int n_in,
                              void* d_out, int out_size, void* d_ws, size_t ws_size,
                              hipStream_t stream) {
  const float* x0  = (const float*)d_in[0];
  const float* x1  = (const float*)d_in[1];
  const float* Wq0 = (const float*)d_in[2];  const float* bq0 = (const float*)d_in[3];
  const float* Wk0 = (const float*)d_in[4];  const float* bk0 = (const float*)d_in[5];
  const float* Wv0 = (const float*)d_in[6];  const float* bv0 = (const float*)d_in[7];
  const float* Wq1 = (const float*)d_in[8];  const float* bq1 = (const float*)d_in[9];
  const float* Wk1 = (const float*)d_in[10]; const float* bk1 = (const float*)d_in[11];
  const float* Wv1 = (const float*)d_in[12]; const float* bv1 = (const float*)d_in[13];

  char* ws = (char*)d_ws;
  __bf16* Qws = (__bf16*)(ws);                         // 2 MB
  __bf16* Kws = (__bf16*)(ws + (size_t)(2 << 20));     // 2 MB
  float*  Vws = (float*)(ws + (size_t)(4 << 20));      // 4 MB
  __bf16* VT  = (__bf16*)(ws + (size_t)(8 << 20));     // 2 MB

  proj_kernel<<<1536, 64, 0, stream>>>(x0, x1, Wq0, bq0, Wk0, bk0, Wv0, bv0,
                                       Wq1, bq1, Wk1, bk1, Wv1, bv1,
                                       Qws, Kws, Vws);
  vsum_kernel<<<4096, 256, 0, stream>>>(Vws, VT);
  attn_kernel<<<256, 64, 2 * 16 * 1024 * sizeof(__bf16), stream>>>(
      Qws, Kws, VT, (float*)d_out);
}
